// VAEAttentionBlock_25795573579973
// MI455X (gfx1250) — compile-verified, matrix operand order checked by reading, not run on hardware
//
#include <hip/hip_runtime.h>
#include <math.h>

// ---------------- types ----------------
typedef _Float16 v16h __attribute__((ext_vector_type(16)));
typedef _Float16 v8h  __attribute__((ext_vector_type(8)));
typedef _Float16 v4h  __attribute__((ext_vector_type(4)));
typedef float    v8f  __attribute__((ext_vector_type(8)));

union H16 { v16h v; v8h h[2]; };

// ---------------- problem constants ----------------
#define BB   4
#define NN   4096            // H*W = 64*64
#define CC   512
#define GG   32
#define CPG  16
#define GN_EPS 1e-5f
#define ATT_SCALE 0.044194173824159216f   // 512^-0.5

// LDS padding to kill bank conflicts (64 banks x 4B)
#define QSTRIDE 520          // 16x512 Q tile rows, padded (+16B)
#define PSTRIDE 136          // 16x128 P tile rows, padded (+16B)

// Load a 16x32 f16 WMMA A/B fragment from a row-major line.
// rowptr points at element (line, k0); hsel = lane>>4 per CDNA5 layout:
//   v[0..7]  = K (hsel? 8..15 : 0..7)
//   v[8..15] = K 16 + (hsel? 8..15 : 0..7)
__device__ __forceinline__ v16h ld_frag(const _Float16* rowptr, int hsel) {
    H16 a;
    a.h[0] = *(const v8h*)(rowptr + hsel * 8);
    a.h[1] = *(const v8h*)(rowptr + 16 + hsel * 8);
    return a.v;
}

__device__ __forceinline__ v8f wmma_f16(v16h a, v16h b, v8f c) {
    return __builtin_amdgcn_wmma_f32_16x16x32_f16(false, a, false, b, (short)0, c, false, false);
}

// ============================================================
// 1) GroupNorm statistics: one block per (b, g); 256 threads
// ============================================================
__global__ void gn_stats_kernel(const float* __restrict__ x, float* __restrict__ stats) {
    const int bg = blockIdx.x;            // b*32 + g
    const int b = bg >> 5, g = bg & 31;
    const float* xb = x + (size_t)b * NN * CC + g * CPG;
    float sum = 0.f, sq = 0.f;
    for (int n = threadIdx.x; n < NN; n += 256) {
        const float4* p = (const float4*)(xb + (size_t)n * CC);
        #pragma unroll
        for (int j = 0; j < 4; ++j) {
            float4 v = p[j];
            sum += v.x + v.y + v.z + v.w;
            sq  += v.x * v.x + v.y * v.y + v.z * v.z + v.w * v.w;
        }
    }
    __shared__ float s_sum[256], s_sq[256];
    s_sum[threadIdx.x] = sum; s_sq[threadIdx.x] = sq;
    __syncthreads();
    for (int s = 128; s > 0; s >>= 1) {
        if (threadIdx.x < s) {
            s_sum[threadIdx.x] += s_sum[threadIdx.x + s];
            s_sq[threadIdx.x]  += s_sq[threadIdx.x + s];
        }
        __syncthreads();
    }
    if (threadIdx.x == 0) {
        const float inv = 1.0f / (float)(NN * CPG);
        float mean = s_sum[0] * inv;
        float var  = s_sq[0] * inv - mean * mean;
        stats[bg]        = mean;
        stats[128 + bg]  = rsqrtf(var + GN_EPS);
    }
}

// ============================================================
// 2) Apply GroupNorm -> f16 activations (4 elems / thread)
// ============================================================
__global__ void gn_apply_kernel(const float* __restrict__ x, const float* __restrict__ stats,
                                const float* __restrict__ gamma, const float* __restrict__ beta,
                                _Float16* __restrict__ h16) {
    const size_t i = ((size_t)blockIdx.x * 256 + threadIdx.x) * 4;
    if (i >= (size_t)BB * NN * CC) return;
    const int c  = (int)(i & (CC - 1));
    const int b  = (int)(i >> 21);                 // / (NN*CC)
    const int bg = b * GG + (c >> 4);
    const float mean = stats[bg], rstd = stats[128 + bg];
    const float4 xv = *(const float4*)(x + i);
    const float4 gv = *(const float4*)(gamma + c);
    const float4 bv = *(const float4*)(beta + c);
    v4h o;
    o[0] = (_Float16)(((xv.x - mean) * rstd) * gv.x + bv.x);
    o[1] = (_Float16)(((xv.y - mean) * rstd) * gv.y + bv.y);
    o[2] = (_Float16)(((xv.z - mean) * rstd) * gv.z + bv.z);
    o[3] = (_Float16)(((xv.w - mean) * rstd) * gv.w + bv.w);
    *(v4h*)(h16 + i) = o;
}

// ============================================================
// 3) Weight transpose -> f16: wt[d*C + c] = w[c*C + d]
// ============================================================
__global__ void wtrans_kernel(const float* __restrict__ w, _Float16* __restrict__ wt) {
    const int id = blockIdx.x * 256 + threadIdx.x;   // d*C + c
    const int d = id >> 9, c = id & (CC - 1);
    wt[id] = (_Float16)w[c * CC + d];
}

// ============================================================
// 4) WMMA GEMM: D[m, d] = A[m, :512] * WtT[d, :512] + bias[d]
//    MODE 0: f16 out, val = scale*(acc+bias)          (Q / K)
//    MODE 1: f16 out transposed [b, d, n]             (V -> Vt)
//    MODE 2: f32 out, val = acc + bias + resid        (proj)
// ============================================================
template <int MODE>
__global__ void gemm512_kernel(const _Float16* __restrict__ A, const _Float16* __restrict__ Wt,
                               const float* __restrict__ bias, void* __restrict__ out,
                               const float* __restrict__ resid, float scale) {
    const int lane = threadIdx.x & 31;
    const int wave = threadIdx.x >> 5;
    const int hsel = lane >> 4;
    const int col  = lane & 15;
    const int mt   = blockIdx.y * 8 + wave;      // 0..1023
    const int m0   = mt * 16;
    const int d0   = blockIdx.x * 16;

    const _Float16* arow = A  + (size_t)(m0 + col) * CC;   // lane's A row
    const _Float16* brow = Wt + (size_t)(d0 + col) * CC;   // lane's B column (transposed W)

    v8f acc = {};
    #pragma unroll
    for (int kt = 0; kt < 16; ++kt) {
        v16h af = ld_frag(arow + kt * 32, hsel);
        v16h bf = ld_frag(brow + kt * 32, hsel);
        acc = wmma_f16(af, bf, acc);
    }

    const float bcol = bias[d0 + col];
    #pragma unroll
    for (int j = 0; j < 8; ++j) {
        const int mrow = m0 + j + 8 * hsel;
        float v;
        if (MODE == 2) v = acc[j] + bcol + resid[(size_t)mrow * CC + d0 + col];
        else           v = scale * (acc[j] + bcol);
        if (MODE == 0) {
            ((_Float16*)out)[(size_t)mrow * CC + d0 + col] = (_Float16)v;
        } else if (MODE == 1) {
            const int b = mrow >> 12, n = mrow & (NN - 1);
            ((_Float16*)out)[((size_t)b * CC + d0 + col) * NN + n] = (_Float16)v;
        } else {
            ((float*)out)[(size_t)mrow * CC + d0 + col] = v;
        }
    }
}

// ============================================================
// 5) Flash-style attention, one block per 16-query tile.
//    128 threads = 4 waves. __launch_bounds__(128,1) gives the
//    compiler the full VGPR budget so the 16 hoisted Q fragments
//    (128 VGPRs) + 8 O accumulators stay in registers (no scratch
//    spills). Pass 1 computes two key tiles/iter to overlap
//    global_load_b128 with WMMA; pass 2 uses 128-key blocks.
// ============================================================
__global__ __launch_bounds__(128, 1)
void attn_kernel(const _Float16* __restrict__ q16, const _Float16* __restrict__ k16,
                 const _Float16* __restrict__ vt16, _Float16* __restrict__ o16) {
    const int lane = threadIdx.x & 31;
    const int wave = threadIdx.x >> 5;          // 0..3
    const int hsel = lane >> 4;
    const int col  = lane & 15;
    const int b  = blockIdx.x >> 8;             // 256 query tiles / batch
    const int m0 = (blockIdx.x & 255) * 16;

    const _Float16* qb  = q16 + (size_t)b * NN * CC;
    const _Float16* kb  = k16 + (size_t)b * NN * CC;
    const _Float16* vtb = vt16 + (size_t)b * CC * NN;
    _Float16* ob        = o16 + (size_t)b * NN * CC;

    __shared__ _Float16 Qlds[16 * QSTRIDE];     // 16x512 Q tile, padded rows
    __shared__ _Float16 Plds[16 * PSTRIDE];     // 16x128 P tile, padded rows
    __shared__ float red[4 * 16];
    __shared__ float rowmaxL[16], rowsumL[16];

    // ---- stage the 16x512 Q tile into LDS (coalesced b128 copies) ----
    for (int i = threadIdx.x; i < 1024; i += 128) {       // 1024 chunks of 8 halfs
        const int row = i >> 6, cidx = (i & 63) * 8;
        *(v8h*)(Qlds + row * QSTRIDE + cidx) =
            *(const v8h*)(qb + (size_t)(m0 + row) * CC + cidx);
    }
    __syncthreads();
    const _Float16* qrow = Qlds + col * QSTRIDE;          // this lane's A row

    // ---- pass 1: global row max over all 4096 keys (2 tiles / iter) ----
    float pm[8];
    #pragma unroll
    for (int j = 0; j < 8; ++j) pm[j] = -3.0e38f;

    for (int kt16 = wave; kt16 < 256; kt16 += 8) {
        const _Float16* krow0 = kb + (size_t)(kt16 * 16 + col) * CC;
        const _Float16* krow1 = kb + (size_t)((kt16 + 4) * 16 + col) * CC;
        v8f s0 = {}, s1 = {};
        #pragma unroll
        for (int ct = 0; ct < 16; ++ct) {
            v16h af = ld_frag(qrow + ct * 32, hsel);     // hoisted to VGPRs
            s0 = wmma_f16(af, ld_frag(krow0 + ct * 32, hsel), s0);
            s1 = wmma_f16(af, ld_frag(krow1 + ct * 32, hsel), s1);
        }
        #pragma unroll
        for (int j = 0; j < 8; ++j) {
            float v = fmaxf(s0[j], s1[j]);
            #pragma unroll
            for (int m = 1; m < 16; m <<= 1) v = fmaxf(v, __shfl_xor(v, m, 32));
            pm[j] = fmaxf(pm[j], v);            // uniform within 16-lane half
        }
    }
    #pragma unroll
    for (int j = 0; j < 8; ++j)
        if (col == 0) red[wave * 16 + j + 8 * hsel] = pm[j];
    __syncthreads();
    if (threadIdx.x < 16) {
        float m = red[threadIdx.x];
        for (int w = 1; w < 4; ++w) m = fmaxf(m, red[w * 16 + threadIdx.x]);
        rowmaxL[threadIdx.x] = m;
    }
    __syncthreads();

    // ---- pass 2: P = exp(S - max), O += P * V^T, rowsum ----
    const int c0 = wave * 128;                  // this wave owns 128 output channels
    v8f oacc[8];
    #pragma unroll
    for (int t = 0; t < 8; ++t) oacc[t] = (v8f){};
    float ps[8];
    #pragma unroll
    for (int j = 0; j < 8; ++j) ps[j] = 0.f;

    for (int kblk = 0; kblk < 32; ++kblk) {     // 128 keys per block-iter
        const int kbase = kblk * 128;
        // each wave computes two 16-key score tiles (keys wave*16 and 64+wave*16)
        {
            const _Float16* krow0 = kb + (size_t)(kbase + wave * 16 + col) * CC;
            const _Float16* krow1 = krow0 + (size_t)64 * CC;
            v8f s0 = {}, s1 = {};
            #pragma unroll
            for (int ct = 0; ct < 16; ++ct) {
                v16h af = ld_frag(qrow + ct * 32, hsel);
                s0 = wmma_f16(af, ld_frag(krow0 + ct * 32, hsel), s0);
                s1 = wmma_f16(af, ld_frag(krow1 + ct * 32, hsel), s1);
            }
            #pragma unroll
            for (int j = 0; j < 8; ++j) {
                const int row = j + 8 * hsel;
                const float rm = rowmaxL[row];
                float e0 = __expf(s0[j] - rm);
                float e1 = __expf(s1[j] - rm);
                ps[j] += e0 + e1;
                Plds[row * PSTRIDE + wave * 16 + col]      = (_Float16)e0;
                Plds[row * PSTRIDE + 64 + wave * 16 + col] = (_Float16)e1;
            }
        }
        __syncthreads();
        // O accumulation: A = P (16x128 from LDS), B = Vt columns
        {
            const _Float16* prow = Plds + col * PSTRIDE;
            v16h pf[4];
            #pragma unroll
            for (int f = 0; f < 4; ++f) pf[f] = ld_frag(prow + f * 32, hsel);
            #pragma unroll
            for (int t = 0; t < 8; ++t) {
                const _Float16* vrow = vtb + (size_t)(c0 + t * 16 + col) * NN + kbase;
                #pragma unroll
                for (int f = 0; f < 4; ++f)
                    oacc[t] = wmma_f16(pf[f], ld_frag(vrow + f * 32, hsel), oacc[t]);
            }
        }
        __syncthreads();
    }

    // ---- merge row sums across lanes and waves ----
    #pragma unroll
    for (int j = 0; j < 8; ++j) {
        float v = ps[j];
        #pragma unroll
        for (int m = 1; m < 16; m <<= 1) v += __shfl_xor(v, m, 32);
        if (col == 0) red[wave * 16 + j + 8 * hsel] = v;
    }
    __syncthreads();
    if (threadIdx.x < 16) {
        float s = red[threadIdx.x];
        for (int w = 1; w < 4; ++w) s += red[w * 16 + threadIdx.x];
        rowsumL[threadIdx.x] = s;
    }
    __syncthreads();

    // ---- normalize and store ----
    #pragma unroll
    for (int t = 0; t < 8; ++t) {
        const float* oa = (const float*)&oacc[t];
        #pragma unroll
        for (int j = 0; j < 8; ++j) {
            const int row = j + 8 * hsel;
            const float v = oa[j] / rowsumL[row];
            ob[(size_t)(m0 + row) * CC + c0 + t * 16 + col] = (_Float16)v;
        }
    }
}

// ============================================================
// host launcher
// ============================================================
extern "C" void kernel_launch(void* const* d_in, const int* in_sizes, int n_in,
                              void* d_out, int out_size, void* d_ws, size_t ws_size,
                              hipStream_t stream) {
    const float* x     = (const float*)d_in[0];
    const float* gamma = (const float*)d_in[1];
    const float* beta  = (const float*)d_in[2];
    const float* wq = (const float*)d_in[3];  const float* bq = (const float*)d_in[4];
    const float* wk = (const float*)d_in[5];  const float* bk = (const float*)d_in[6];
    const float* wv = (const float*)d_in[7];  const float* bv = (const float*)d_in[8];
    const float* wp = (const float*)d_in[9];  const float* bp = (const float*)d_in[10];
    float* out = (float*)d_out;

    // workspace carve-out
    char* ws = (char*)d_ws;
    size_t off = 0;
    auto carve = [&](size_t bytes) -> char* {
        char* p = ws + off;
        off = (off + bytes + 255) & ~(size_t)255;
        return p;
    };
    const size_t act_h = (size_t)BB * NN * CC * sizeof(_Float16);   // 16 MB
    float*    stats = (float*)   carve(2 * 128 * sizeof(float));
    _Float16* h16   = (_Float16*)carve(act_h);
    _Float16* wqT   = (_Float16*)carve((size_t)CC * CC * sizeof(_Float16));
    _Float16* wkT   = (_Float16*)carve((size_t)CC * CC * sizeof(_Float16));
    _Float16* wvT   = (_Float16*)carve((size_t)CC * CC * sizeof(_Float16));
    _Float16* wpT   = (_Float16*)carve((size_t)CC * CC * sizeof(_Float16));
    _Float16* q16   = (_Float16*)carve(act_h);
    _Float16* k16   = (_Float16*)carve(act_h);
    _Float16* vt16  = (_Float16*)carve(act_h);
    _Float16* o16   = (_Float16*)carve(act_h);
    (void)ws_size; (void)n_in; (void)in_sizes; (void)out_size;

    // 1) GroupNorm stats
    gn_stats_kernel<<<BB * GG, 256, 0, stream>>>(x, stats);
    // 2) apply GN -> f16
    gn_apply_kernel<<<(BB * NN * CC) / (256 * 4), 256, 0, stream>>>(x, stats, gamma, beta, h16);
    // 3) transpose weights -> f16
    wtrans_kernel<<<(CC * CC) / 256, 256, 0, stream>>>(wq, wqT);
    wtrans_kernel<<<(CC * CC) / 256, 256, 0, stream>>>(wk, wkT);
    wtrans_kernel<<<(CC * CC) / 256, 256, 0, stream>>>(wv, wvT);
    wtrans_kernel<<<(CC * CC) / 256, 256, 0, stream>>>(wp, wpT);
    // 4) QKV projections (WMMA); Q pre-scaled by 1/sqrt(C); V stored transposed
    dim3 ggrid(CC / 16, (BB * NN) / (16 * 8));
    gemm512_kernel<0><<<ggrid, 256, 0, stream>>>(h16, wqT, bq, (void*)q16, nullptr, ATT_SCALE);
    gemm512_kernel<0><<<ggrid, 256, 0, stream>>>(h16, wkT, bk, (void*)k16, nullptr, 1.0f);
    gemm512_kernel<1><<<ggrid, 256, 0, stream>>>(h16, wvT, bv, (void*)vt16, nullptr, 1.0f);
    // 5) attention
    attn_kernel<<<BB * (NN / 16), 128, 0, stream>>>(q16, k16, vt16, o16);
    // 6) projection + residual -> f32 output
    gemm512_kernel<2><<<ggrid, 256, 0, stream>>>(o16, wpT, bp, (void*)out, x, 1.0f);
}
